// QuantizationLayer_37323265802266
// MI455X (gfx1250) — matrix operand order repaired and checked by reading, not compiled, not run on hardware
//
#include <hip/hip_runtime.h>
#include <hip/hip_bf16.h>

// ---------------------------------------------------------------------------
// Event-camera voxelization with a per-scalar value MLP (1->100->100->1).
// Layer-2 (100x100) executed on CDNA5 WMMA f16 tensor cores with f32 accum.
// ---------------------------------------------------------------------------

typedef __attribute__((ext_vector_type(16))) _Float16 v16h;
typedef __attribute__((ext_vector_type(8)))  float    v8f;

#define B_  16
#define C_  9
#define H_  180
#define W_  240
#define HWSZ (H_ * W_)

// hidden padded to 128 (4 K-tiles of 32), output cols padded to 112 (7 N-tiles)
#define KT_ 4
#define NT_ 7
#define NFRAG (KT_ * NT_)          // 28 B fragments
#define FRAG_HALVES (32 * 16)      // per fragment: 32 lanes x 16 halves

// workspace byte offsets
#define WS_TMAX 0                  // 16 ints (float bits)
#define WS_W1P  64                 // 128 f32
#define WS_B1P  576                // 128 f32
#define WS_B2P  1088               // 128 f32 (100 used)
#define WS_W3P  1600               // 128 f32 (100 used)
#define WS_WB   2112               // 28*512 halves = 28672 B

// ---------------------------------------------------------------------------
__global__ void zero_kernel(float4* __restrict__ out, int n4) {
    int stride = gridDim.x * blockDim.x;
    for (int i = blockIdx.x * blockDim.x + threadIdx.x; i < n4; i += stride)
        out[i] = make_float4(0.f, 0.f, 0.f, 0.f);
}

// ---------------------------------------------------------------------------
// Pack params: pad w1/b1/b2/w3 to 128, zero tmax, and lay w2 (f32, [100,100]
// row-major (in,out)) into WMMA B-fragment order as f16:
//   fragment (kt,nt), lane l, half h  ->  K = kt*32 + (l<16?0:8) + (h&7) + (h<8?0:16)
//                                         N = nt*16 + (l&15)
__global__ void prep_kernel(const float* __restrict__ w1, const float* __restrict__ b1,
                            const float* __restrict__ w2, const float* __restrict__ b2,
                            const float* __restrict__ w3, char* __restrict__ ws) {
    int tid = threadIdx.x;
    if (tid < 16) ((int*)(ws + WS_TMAX))[tid] = 0;   // t > 0 so int-max == float-max
    float* w1p = (float*)(ws + WS_W1P);
    float* b1p = (float*)(ws + WS_B1P);
    float* b2p = (float*)(ws + WS_B2P);
    float* w3p = (float*)(ws + WS_W3P);
    for (int i = tid; i < 128; i += 256) {
        w1p[i] = (i < 100) ? w1[i] : 0.f;
        b1p[i] = (i < 100) ? b1[i] : 0.f;
        b2p[i] = (i < 100) ? b2[i] : 0.f;
        w3p[i] = (i < 100) ? w3[i] : 0.f;
    }
    _Float16* wB = (_Float16*)(ws + WS_WB);
    for (int idx = tid; idx < NFRAG * FRAG_HALVES; idx += 256) {
        int h    = idx & 15;
        int lane = (idx >> 4) & 31;
        int frag = idx >> 9;
        int kt = frag / NT_, nt = frag % NT_;
        int K  = kt * 32 + ((lane & 16) ? 8 : 0) + (h & 7) + ((h & 8) ? 16 : 0);
        int Nn = nt * 16 + (lane & 15);
        float v = (K < 100 && Nn < 100) ? w2[K * 100 + Nn] : 0.f;
        wB[idx] = (_Float16)v;
    }
}

// ---------------------------------------------------------------------------
__global__ void tmax_kernel(const float* __restrict__ t, const int* __restrict__ bi,
                            char* __restrict__ ws, int n) {
    int* tm = (int*)(ws + WS_TMAX);
    int stride = gridDim.x * blockDim.x;
    for (int i = blockIdx.x * blockDim.x + threadIdx.x; i < n; i += stride)
        atomicMax(&tm[bi[i]], __float_as_int(t[i]));   // t in (0,1]: bits are order-preserving
}

// ---------------------------------------------------------------------------
// Main kernel: each wave handles 16 events x 9 bins.
//   A fragment (16x32 f16, ISA layout): lane l -> row M = l&15;
//     kbase = kt*32 + (l<16?0:8); halves 0..7 -> K=kbase+0..7,
//     halves 8..15 -> K=kbase+16..23.
//   D tile (16x16 f32): VGPR r, lane l -> (M = r + (l<16?0:8), N = l&15).
__global__ void __launch_bounds__(256)
voxel_mlp_kernel(const float* __restrict__ t,  const int* __restrict__ xi,
                 const int* __restrict__ yi,   const int* __restrict__ pi,
                 const int* __restrict__ bi,   const float* __restrict__ b3,
                 const char* __restrict__ ws,  float* __restrict__ vox, int n) {
    __shared__ __align__(32) _Float16 s_wB[NFRAG * FRAG_HALVES];
    __shared__ __align__(16) float s_w1p[128], s_b1p[128], s_b2p[128], s_w3p[128];
    __shared__ float s_tinv[16];

    const int tid = threadIdx.x;
    {   // stage packed weights + params into LDS
        const int* src = (const int*)(ws + WS_WB);
        int* dst = (int*)s_wB;
        for (int i = tid; i < NFRAG * FRAG_HALVES / 2; i += 256) dst[i] = src[i];
        const float* w1p = (const float*)(ws + WS_W1P);
        const float* b1p = (const float*)(ws + WS_B1P);
        const float* b2p = (const float*)(ws + WS_B2P);
        const float* w3p = (const float*)(ws + WS_W3P);
        for (int i = tid; i < 128; i += 256) {
            s_w1p[i] = w1p[i]; s_b1p[i] = b1p[i];
            s_b2p[i] = b2p[i]; s_w3p[i] = w3p[i];
        }
        if (tid < 16) s_tinv[tid] = 1.0f / __int_as_float(((const int*)(ws + WS_TMAX))[tid]);
    }
    __syncthreads();

    const int lane  = tid & 31;
    const int gwave = blockIdx.x * (blockDim.x >> 5) + (tid >> 5);
    const int nwav  = gridDim.x * (blockDim.x >> 5);
    const int ntile = (n + 15) >> 4;
    const float b3v = b3[0];
    const int m   = lane & 15;
    const int kb0 = (lane & 16) ? 8 : 0;
    // lanes 0..7 write rows 0..7, lanes 16..23 write rows 8..15
    const int row = (lane < 16) ? lane : (lane - 8);

    for (int tile = gwave; tile < ntile; tile += nwav) {
        const int base = tile << 4;
        int em = base + m;          if (em >= n) em = n - 1;
        const float tnm = t[em] * s_tinv[bi[em]];        // normalized ts for row m

        bool doAt = ((lane < 8) || (lane >= 16 && lane < 24));
        int er = base + (row & 15);
        doAt = doAt && (er < n);
        if (er >= n) er = n - 1;
        const float tnr = t[er] * s_tinv[bi[er]];
        const int idxb  = ((bi[er] * 2 + pi[er]) * C_) * HWSZ + yi[er] * W_ + xi[er];

        for (int c = 0; c < C_; ++c) {
            const float s = tnm - (float)c * 0.125f;     // tn - c/(C-1)

            // ---- layer 1 straight into A fragments (f32 VALU -> f16 pack)
            v16h afr[KT_];
            #pragma unroll
            for (int kt = 0; kt < KT_; ++kt) {
                const int kbase = kt * 32 + kb0;         // multiple of 8
                float wv[16], bv[16];
                *(float4*)&wv[0]  = *(const float4*)&s_w1p[kbase];
                *(float4*)&wv[4]  = *(const float4*)&s_w1p[kbase + 4];
                *(float4*)&wv[8]  = *(const float4*)&s_w1p[kbase + 16];
                *(float4*)&wv[12] = *(const float4*)&s_w1p[kbase + 20];
                *(float4*)&bv[0]  = *(const float4*)&s_b1p[kbase];
                *(float4*)&bv[4]  = *(const float4*)&s_b1p[kbase + 4];
                *(float4*)&bv[8]  = *(const float4*)&s_b1p[kbase + 16];
                *(float4*)&bv[12] = *(const float4*)&s_b1p[kbase + 20];
                #pragma unroll
                for (int h = 0; h < 16; ++h) {
                    float x = fmaf(s, wv[h], bv[h]);
                    x = fmaxf(x, 0.1f * x);              // LeakyReLU(0.1)
                    afr[kt][h] = (_Float16)x;
                }
            }

            // ---- layer 2 on WMMA + fused layer-3 partials
            v8f rowsum = {};
            #pragma unroll
            for (int nt = 0; nt < NT_; ++nt) {
                v8f acc = {};
                #pragma unroll
                for (int kt = 0; kt < KT_; ++kt) {
                    v16h bf = *(const v16h*)&s_wB[(((kt * NT_) + nt) * 32 + lane) * 16];
                    acc = __builtin_amdgcn_wmma_f32_16x16x32_f16(
                        false, afr[kt], false, bf, (short)0, acc, false, false);
                }
                const float b2v = s_b2p[nt * 16 + m];
                const float w3v = s_w3p[nt * 16 + m];
                #pragma unroll
                for (int r = 0; r < 8; ++r) {
                    float h2 = acc[r] + b2v;
                    h2 = fmaxf(h2, 0.1f * h2);
                    rowsum[r] = fmaf(h2, w3v, rowsum[r]);
                }
            }

            // ---- reduce layer-3 dot over the 16 lanes of each half-wave
            float red[8];
            #pragma unroll
            for (int r = 0; r < 8; ++r) {
                float v = rowsum[r];
                v += __shfl_xor(v, 1);
                v += __shfl_xor(v, 2);
                v += __shfl_xor(v, 4);
                v += __shfl_xor(v, 8);
                red[r] = v;
            }
            const int rsel = lane & 7;
            float mysum = red[0];
            #pragma unroll
            for (int rr = 1; rr < 8; ++rr) if (rsel == rr) mysum = red[rr];

            if (doAt)
                atomicAdd(&vox[idxb + c * HWSZ], tnr * (mysum + b3v));
        }
    }
}

// ---------------------------------------------------------------------------
extern "C" void kernel_launch(void* const* d_in, const int* in_sizes, int n_in,
                              void* d_out, int out_size, void* d_ws, size_t ws_size,
                              hipStream_t stream) {
    const float* t  = (const float*)d_in[0];
    const int*   xi = (const int*)  d_in[1];
    const int*   yi = (const int*)  d_in[2];
    const int*   pi = (const int*)  d_in[3];
    const int*   bi = (const int*)  d_in[4];
    const float* w1 = (const float*)d_in[5];
    const float* b1 = (const float*)d_in[6];
    const float* w2 = (const float*)d_in[7];
    const float* b2 = (const float*)d_in[8];
    const float* w3 = (const float*)d_in[9];
    const float* b3 = (const float*)d_in[10];
    float* vox = (float*)d_out;
    char*  ws  = (char*)d_ws;
    const int n = in_sizes[0];

    // 1) zero the 50 MB voxel accumulator (poisoned by harness; atomics need 0)
    int n4 = out_size / 4;
    zero_kernel<<<4096, 256, 0, stream>>>((float4*)d_out, n4);
    // 2) pack weights into WMMA fragment order, pad params, zero tmax
    prep_kernel<<<1, 256, 0, stream>>>(w1, b1, w2, b2, w3, ws);
    // 3) per-batch segment max of t via int-bit atomicMax
    tmax_kernel<<<2048, 256, 0, stream>>>(t, bi, ws, n);
    // 4) main WMMA voxelization
    voxel_mlp_kernel<<<1024, 256, 0, stream>>>(t, xi, yi, pi, bi, b3, ws, vox, n);
}
